// S2VSingle_53343493816571
// MI455X (gfx1250) — matrix operand order repaired and checked by reading, not compiled, not run on hardware
//
#include <hip/hip_runtime.h>
#include <hip/hip_bf16.h>

// ---------------------------------------------------------------------------
// structure2vec GNN forward for MI455X (gfx1250), wave32 + WMMA f16.
//  - edge-feature pooling done ONCE (linear commutes with segment_sum)
//  - GEMMs use v_wmma_f32_16x16x32_f16 with fragment-swizzled LDS staging so
//    every fragment is a single contiguous 32B (2x ds_load_b128) LDS read.
//  - each wave computes a 32x64 tile: B fragments are reused by two A tiles.
// ---------------------------------------------------------------------------

typedef _Float16 half_t;
typedef half_t v16h __attribute__((ext_vector_type(16)));
typedef float  v8f  __attribute__((ext_vector_type(8)));

#define D 64
#define NF 128
#define EF 32
#define LV 3
#define NG 64
#define BN_EPS 1e-5f

// --------------------------- utility fills ---------------------------------
__global__ void fill_f32_kernel(float* p, float v, long n) {
    long i = (long)blockIdx.x * blockDim.x + threadIdx.x;
    if (i < n) p[i] = v;
}
__global__ void fill_i32_kernel(int* p, int v, long n) {
    long i = (long)blockIdx.x * blockDim.x + threadIdx.x;
    if (i < n) p[i] = v;
}

// ------------------- edge feature pooling (once, not 4x) -------------------
__global__ void edge_pool_kernel(const float* __restrict__ edge_feat,
                                 const int* __restrict__ edge_to,
                                 float* __restrict__ pooled_e,
                                 float* __restrict__ deg, long E) {
    long t = (long)blockIdx.x * blockDim.x + threadIdx.x;  // E*8 threads
    long e = t >> 3;
    int  c = (int)(t & 7) * 4;
    if (e >= E) return;
    int dst = edge_to[e];
    const float4 v = *(const float4*)(edge_feat + e * EF + c);
    float* base = pooled_e + (size_t)dst * EF + c;
    atomicAdd(base + 0, v.x);
    atomicAdd(base + 1, v.y);
    atomicAdd(base + 2, v.z);
    atomicAdd(base + 3, v.w);
    if (c == 0) atomicAdd(deg + dst, 1.0f);
}

// ----------------- sparse message passing: agg[to] += h[from] --------------
__global__ void spmm_scatter_kernel(const float* __restrict__ h,
                                    const int* __restrict__ edge_from,
                                    const int* __restrict__ edge_to,
                                    float* __restrict__ agg, long E) {
    long t = (long)blockIdx.x * blockDim.x + threadIdx.x;  // E*16 threads
    long e = t >> 4;
    int  c = (int)(t & 15) * 4;
    if (e >= E) return;
    int src = edge_from[e];
    int dst = edge_to[e];
    const float4 v = *(const float4*)(h + (size_t)src * D + c);
    float* base = agg + (size_t)dst * D + c;
    atomicAdd(base + 0, v.x);
    atomicAdd(base + 1, v.y);
    atomicAdd(base + 2, v.z);
    atomicAdd(base + 3, v.w);
}

// ---------------- fragment swizzle helpers (ISA 7.12.2 layouts) ------------
// A fragment, lane L, half h: row = (L&15), k = (h<8 ? h : h+8) + 8*(L>>4)
__device__ inline int a_frag_k(int L, int h) {
    return ((h < 8) ? h : h + 8) + ((L >> 4) << 3);
}
// B fragment, lane L, half h: col = (L&15), k = (h<8 ? h : h+8) + 8*(L>>4)
__device__ inline int b_frag_k(int L, int h) {
    return ((h < 8) ? h : h + 8) + ((L >> 4) << 3);
}

__device__ inline v8f wmma_step(const half_t* aFrag, const half_t* bFrag,
                                v8f acc) {
    v16h a = *(const v16h*)aFrag;
    v16h b = *(const v16h*)bFrag;
    return __builtin_amdgcn_wmma_f32_16x16x32_f16(false, a, false, b, (short)0,
                                                  acc, false, false);
}

// ------------------- dense GEMM: out = A(NxK) @ B(Kx64) + eps --------------
// 128 threads = 4 waves; block covers 128 rows (8 row-tiles of 16, 2/wave).
// epilogue: + rowscale[i]*bias[j] (rowscale==null -> 1) + addend[i][j], relu
template <int K>
__global__ __launch_bounds__(128) void gemm_kernel(
    const float* __restrict__ A, const float* __restrict__ B,
    const float* __restrict__ bias, const float* __restrict__ rowscale,
    const float* __restrict__ addend, float* __restrict__ out, int N,
    int relu) {
    constexpr int KT = K / 32;
    __shared__ half_t AsF[8][KT][32][16];  // [rowtile][ktile][lane][half]
    __shared__ half_t BsF[KT][4][32][16];  // [ktile][ntile][lane][half]
    const int tid = threadIdx.x;
    const int wave = tid >> 5;
    const int lane = tid & 31;
    const int rowBase = blockIdx.x * 128;

    // stage B fragments (f32 -> f16, fragment-major so loads are v16h)
    for (int i = tid; i < KT * 4 * 32 * 16; i += 128) {
        int h = i & 15;
        int L = (i >> 4) & 31;
        int nt = (i >> 9) & 3;
        int kt = i >> 11;
        int kk = kt * 32 + b_frag_k(L, h);
        int col = nt * 16 + (L & 15);
        ((half_t*)BsF)[i] = (half_t)B[kk * D + col];
    }
    // stage A fragments (zero-padded beyond N)
    for (int i = tid; i < 8 * KT * 32 * 16; i += 128) {
        int h = i & 15;
        int L = (i >> 4) & 31;
        int kt = (i >> 9) % KT;
        int rt = (i >> 9) / KT;
        int row = rowBase + rt * 16 + (L & 15);
        int kk = kt * 32 + a_frag_k(L, h);
        ((half_t*)AsF)[i] =
            (row < N) ? (half_t)A[(size_t)row * K + kk] : (half_t)0;
    }
    __syncthreads();

    const int rt0 = wave * 2;
    if (rowBase + rt0 * 16 >= N) return;  // wave-uniform; no barriers follow

    v8f acc[2][4];
#pragma unroll
    for (int s = 0; s < 2; ++s)
#pragma unroll
        for (int nt = 0; nt < 4; ++nt)
#pragma unroll
            for (int i = 0; i < 8; ++i) acc[s][nt][i] = 0.0f;

#pragma unroll
    for (int kt = 0; kt < KT; ++kt) {
        const half_t* a0 = AsF[rt0 + 0][kt][lane];
        const half_t* a1 = AsF[rt0 + 1][kt][lane];
#pragma unroll
        for (int nt = 0; nt < 4; ++nt) {
            const half_t* b = BsF[kt][nt][lane];
            acc[0][nt] = wmma_step(a0, b, acc[0][nt]);
            acc[1][nt] = wmma_step(a1, b, acc[1][nt]);
        }
    }

    const int lm = lane & 15;
    const int hi = lane >> 4;
#pragma unroll
    for (int s = 0; s < 2; ++s) {
#pragma unroll
        for (int nt = 0; nt < 4; ++nt) {
            int col = nt * 16 + lm;
#pragma unroll
            for (int r = 0; r < 8; ++r) {
                int row = rowBase + (rt0 + s) * 16 + hi * 8 + r;
                if (row < N) {
                    float v = acc[s][nt][r];
                    if (bias) v += (rowscale ? rowscale[row] : 1.0f) * bias[col];
                    if (addend) v += addend[(size_t)row * D + col];
                    if (relu) v = v > 0.0f ? v : 0.0f;
                    out[(size_t)row * D + col] = v;
                }
            }
        }
    }
}

// ---------------- embedding gather add: x[i] += emb[idx[i]] ---------------
__global__ void add_emb_kernel(float* __restrict__ x,
                               const float* __restrict__ emb,
                               const int* __restrict__ idx, long N) {
    long t = (long)blockIdx.x * blockDim.x + threadIdx.x;
    if (t >= N * D) return;
    long row = t >> 6;
    int col = (int)(t & 63);
    x[t] += emb[(size_t)idx[row] * D + col];
}

// ---------------- BatchNorm (training-mode biased stats) -------------------
__global__ __launch_bounds__(256) void bn_stats_kernel(
    const float* __restrict__ x, float* __restrict__ stats, long N) {
    __shared__ float s[2 * D];
    int tid = threadIdx.x;
    if (tid < 2 * D) s[tid] = 0.0f;
    __syncthreads();
    long total = N * D;
    for (long i = (long)blockIdx.x * blockDim.x + tid; i < total;
         i += (long)gridDim.x * blockDim.x) {
        float v = x[i];
        int col = (int)(i & 63);
        atomicAdd(&s[col], v);
        atomicAdd(&s[D + col], v * v);
    }
    __syncthreads();
    if (tid < 2 * D) atomicAdd(&stats[tid], s[tid]);
}

__global__ void bn_apply_kernel(const float* __restrict__ x,
                                const float* __restrict__ stats,
                                const float* __restrict__ g,
                                const float* __restrict__ b,
                                float* __restrict__ out, long N) {
    long t = (long)blockIdx.x * blockDim.x + threadIdx.x;
    if (t >= N * D) return;
    int col = (int)(t & 63);
    float nf = (float)N;
    float m = stats[col] / nf;
    float var = stats[D + col] / nf - m * m;
    out[t] = (x[t] - m) * rsqrtf(var + BN_EPS) * g[col] + b[col];
}

// ---------------- segment-max readout via ordered-int atomicMax ------------
__device__ inline int f32_to_ordered(float f) {
    int i = __float_as_int(f);
    return (i >= 0) ? i : (i ^ 0x7fffffff);
}
__device__ inline float ordered_to_f32(int k) {
    return (k >= 0) ? __int_as_float(k) : __int_as_float(k ^ 0x7fffffff);
}

__global__ void segmax_kernel(const float* __restrict__ h,
                              const int* __restrict__ g_idx,
                              int* __restrict__ keys, long N) {
    long t = (long)blockIdx.x * blockDim.x + threadIdx.x;
    if (t >= N * D) return;
    long row = t >> 6;
    int col = (int)(t & 63);
    int g = g_idx[row];
    atomicMax(&keys[g * D + col], f32_to_ordered(h[t]));
}

// ---------------- final: out = relu(pooled(64x64) @ W + b), 1 block --------
__global__ __launch_bounds__(128) void readout_kernel(
    const int* __restrict__ keys, const float* __restrict__ W,
    const float* __restrict__ bias, float* __restrict__ out) {
    __shared__ half_t PF[4][2][32][16];  // [rowtile][ktile][lane][half]
    __shared__ half_t WF[2][4][32][16];  // [ktile][ntile][lane][half]
    const int tid = threadIdx.x;
    const int wave = tid >> 5;
    const int lane = tid & 31;

    for (int i = tid; i < 4 * 2 * 32 * 16; i += 128) {
        int h = i & 15;
        int L = (i >> 4) & 31;
        int kt = (i >> 9) & 1;
        int rt = i >> 10;
        int row = rt * 16 + (L & 15);
        int kk = kt * 32 + a_frag_k(L, h);
        ((half_t*)PF)[i] = (half_t)ordered_to_f32(keys[row * D + kk]);
    }
    for (int i = tid; i < 2 * 4 * 32 * 16; i += 128) {
        int h = i & 15;
        int L = (i >> 4) & 31;
        int nt = (i >> 9) & 3;
        int kt = i >> 11;
        int kk = kt * 32 + b_frag_k(L, h);
        int col = nt * 16 + (L & 15);
        ((half_t*)WF)[i] = (half_t)W[kk * D + col];
    }
    __syncthreads();

    v8f acc[4];
#pragma unroll
    for (int nt = 0; nt < 4; ++nt)
#pragma unroll
        for (int i = 0; i < 8; ++i) acc[nt][i] = 0.0f;
#pragma unroll
    for (int kt = 0; kt < 2; ++kt) {
        const half_t* a = PF[wave][kt][lane];
#pragma unroll
        for (int nt = 0; nt < 4; ++nt)
            acc[nt] = wmma_step(a, WF[kt][nt][lane], acc[nt]);
    }
    const int lm = lane & 15;
    const int hi = lane >> 4;
#pragma unroll
    for (int nt = 0; nt < 4; ++nt) {
        int col = nt * 16 + lm;
#pragma unroll
        for (int r = 0; r < 8; ++r) {
            int row = wave * 16 + hi * 8 + r;
            float v = acc[nt][r] + bias[col];
            out[row * D + col] = v > 0.0f ? v : 0.0f;
        }
    }
}

// ---------------------------------------------------------------------------
static inline int gb(long n, int bs) { return (int)((n + bs - 1) / bs); }

extern "C" void kernel_launch(void* const* d_in, const int* in_sizes, int n_in,
                              void* d_out, int out_size, void* d_ws,
                              size_t ws_size, hipStream_t stream) {
    (void)n_in; (void)out_size; (void)ws_size;
    const float* node_feat = (const float*)d_in[0];
    const float* edge_feat = (const float*)d_in[1];
    const float* w_n2l_W   = (const float*)d_in[2];
    const float* w_n2l_b   = (const float*)d_in[3];
    const float* emb       = (const float*)d_in[4];
    const float* w_e2l_W   = (const float*)d_in[5];   // (LV+1, EF, D)
    const float* w_e2l_b   = (const float*)d_in[6];   // (LV+1, D)
    const float* conv_W    = (const float*)d_in[7];   // (LV, D, D)
    const float* conv_b    = (const float*)d_in[8];
    const float* l2_W      = (const float*)d_in[9];
    const float* l2_b      = (const float*)d_in[10];
    const float* msg_g     = (const float*)d_in[11];
    const float* msg_b     = (const float*)d_in[12];
    const float* hid_g     = (const float*)d_in[13];
    const float* hid_b     = (const float*)d_in[14];
    const float* out_W     = (const float*)d_in[15];
    const float* out_b     = (const float*)d_in[16];
    const int* edge_from   = (const int*)d_in[17];
    const int* edge_to     = (const int*)d_in[18];
    const int* g_idx       = (const int*)d_in[19];
    const int* nv_idx      = (const int*)d_in[20];

    const long N = in_sizes[19];  // 100000
    const long E = in_sizes[17];  // 1600000

    // workspace layout (floats); all offsets 16B aligned
    float* ws = (float*)d_ws;
    float* pooled_e = ws;                         // N*EF
    float* deg      = pooled_e + N * EF;          // N
    float* stats    = deg + ((N + 3) & ~3L);      // 7*128
    int*   keys     = (int*)(stats + 7 * 128);    // NG*D
    float* h        = (float*)(keys + ((NG * D + 3) & ~3));  // N*D
    float* tmp1     = h + N * D;                  // N*D
    float* tmp2     = tmp1 + N * D;               // N*D
    float* agg      = tmp2 + N * D;               // N*D

    const long ND = N * D;
    const int gemmBlocks = gb(N, 128);
    const int NEG_KEY = 0x80800000;  // ordered encoding of -FLT_MAX

    // init accumulators (deterministic every call)
    fill_f32_kernel<<<gb(N * EF, 256), 256, 0, stream>>>(pooled_e, 0.f, N * EF);
    fill_f32_kernel<<<gb(N, 256), 256, 0, stream>>>(deg, 0.f, N);
    fill_f32_kernel<<<gb(7 * 128, 256), 256, 0, stream>>>(stats, 0.f, 7 * 128);
    fill_i32_kernel<<<gb(NG * D, 256), 256, 0, stream>>>(keys, NEG_KEY, NG * D);

    // pool edge features ONCE (replaces 4 E-sized GEMM+scatter passes)
    edge_pool_kernel<<<gb(E * 8, 256), 256, 0, stream>>>(edge_feat, edge_to,
                                                         pooled_e, deg, E);

    // h0 = relu(node_feat@Wn2l + bn2l + emb[idx] + pooled_e@We2l0 + deg*be2l0)
    gemm_kernel<NF><<<gemmBlocks, 128, 0, stream>>>(
        node_feat, w_n2l_W, w_n2l_b, nullptr, nullptr, tmp1, (int)N, 0);
    add_emb_kernel<<<gb(ND, 256), 256, 0, stream>>>(tmp1, emb, nv_idx, N);
    gemm_kernel<EF><<<gemmBlocks, 128, 0, stream>>>(
        pooled_e, w_e2l_W, w_e2l_b, deg, tmp1, tmp2, (int)N, 1);
    bn_stats_kernel<<<512, 256, 0, stream>>>(tmp2, stats, N);
    bn_apply_kernel<<<gb(ND, 256), 256, 0, stream>>>(tmp2, stats, msg_g, msg_b,
                                                     h, N);

    for (int lv = 0; lv < LV; ++lv) {
        float* stHid = stats + (1 + 2 * lv) * 128;
        float* stMsg = stats + (2 + 2 * lv) * 128;
        // agg = scatter_add(h[from] -> to)
        fill_f32_kernel<<<gb(ND, 256), 256, 0, stream>>>(agg, 0.f, ND);
        spmm_scatter_kernel<<<gb(E * 16, 256), 256, 0, stream>>>(
            h, edge_from, edge_to, agg, E);
        // node_linear = agg @ conv_W + conv_b
        gemm_kernel<D><<<gemmBlocks, 128, 0, stream>>>(
            agg, conv_W + (size_t)lv * D * D, conv_b + lv * D, nullptr,
            nullptr, tmp1, (int)N, 0);
        // merged = relu(node_linear + pooled_e@We2l[lv+1] + deg*be2l[lv+1])
        gemm_kernel<EF><<<gemmBlocks, 128, 0, stream>>>(
            pooled_e, w_e2l_W + (size_t)(lv + 1) * EF * D,
            w_e2l_b + (lv + 1) * D, deg, tmp1, tmp2, (int)N, 1);
        bn_stats_kernel<<<512, 256, 0, stream>>>(tmp2, stHid, N);
        bn_apply_kernel<<<gb(ND, 256), 256, 0, stream>>>(
            tmp2, stHid, hid_g + lv * D, hid_b + lv * D, tmp1, N);
        // h = bn(relu(merged @ l2_W + l2_b + h))
        gemm_kernel<D><<<gemmBlocks, 128, 0, stream>>>(
            tmp1, l2_W + (size_t)lv * D * D, l2_b + lv * D, nullptr, h, tmp2,
            (int)N, 1);
        bn_stats_kernel<<<512, 256, 0, stream>>>(tmp2, stMsg, N);
        bn_apply_kernel<<<gb(ND, 256), 256, 0, stream>>>(
            tmp2, stMsg, msg_g + (lv + 1) * D, msg_b + (lv + 1) * D, h, N);
    }

    // readout: per-graph segment max, then relu(pooled @ out_W + out_b)
    segmax_kernel<<<gb(ND, 256), 256, 0, stream>>>(h, g_idx, keys, N);
    readout_kernel<<<1, 128, 0, stream>>>(keys, out_W, out_b, (float*)d_out);
}
